// RMSNorm_29626684408043
// MI455X (gfx1250) — matrix operand order repaired and checked by reading, not compiled, not run on hardware
//
#include <hip/hip_runtime.h>

// RMSNorm for [4, 8192, 4096] fp16 on MI455X (gfx1250, wave32).
//
// Bandwidth-bound: 536 MB mandatory HBM traffic -> ~23us floor at 23.3 TB/s.
// Strategy:
//   * block = 512 threads = 16 waves, handles 16 rows (32768 waves total ->
//     ample memory-level parallelism for Little's-law ~14MB in flight).
//   * wave w owns K-slice [w*256, w*256+256) of all 16 rows, kept register
//     resident (8 x v16h per lane) so the scale pass re-reads NO global memory.
//   * sum(x^2) per row via V_WMMA_F32_16X16X32_F16: D = A*B with A = x tile,
//     B = x^T tile => diag(D) = per-row sum of squares, accumulated in f32
//     (matches reference's fp32 upcast). Both A and B layouts are direct
//     b128 loads from row-major memory (B: lane=column of x^T = row of x,
//     16 contiguous K per lane; A: same rows, same 16B segments/lane —
//     the A view re-reads lines just fetched by the B view, hitting L0).
//   * wave partial diagonals combined through LDS ds_add_f32 (16 floats),
//     rsqrt + 1 Newton iteration, then scale + 32B stores per lane.
//   * stores are non-temporal: output is streamed write-once, keeping it out
//     of L2 preserves the 192MB cache for in-flight input lines.

typedef _Float16 v8h  __attribute__((ext_vector_type(8)));
typedef _Float16 v16h __attribute__((ext_vector_type(16)));
typedef float    v8f  __attribute__((ext_vector_type(8)));

#define MEAN_DIM        4096
#define EPS             1e-5f
#define ROWS_PER_BLOCK  16
#define WAVES_PER_BLOCK 16
#define THREADS         (WAVES_PER_BLOCK * 32)
#define K_PER_WAVE      (MEAN_DIM / WAVES_PER_BLOCK)  // 256
#define CHUNKS          (K_PER_WAVE / 32)             // 8 WMMA K-steps

__global__ __launch_bounds__(THREADS)
void rmsnorm_wmma_kernel(const _Float16* __restrict__ X,
                         _Float16* __restrict__ O) {
    __shared__ float ssum[ROWS_PER_BLOCK];

    const int t    = threadIdx.x;
    const int wave = t >> 5;
    const int lane = t & 31;
    const int row  = lane & 15;   // which of the block's 16 rows this lane maps to
    const int half = lane >> 4;   // 0: lanes 0-15, 1: lanes 16-31

    if (t < ROWS_PER_BLOCK) ssum[t] = 0.0f;
    __syncthreads();

    const size_t rowbase = ((size_t)blockIdx.x * ROWS_PER_BLOCK + row) * (size_t)MEAN_DIM;
    const _Float16* xr = X + rowbase + (size_t)wave * K_PER_WAVE;

    v16h xb[CHUNKS];   // register-resident B-layout data (reused for the store pass)
    v8f  acc = {};     // 16x16 f32 Gram accumulator

#pragma unroll
    for (int j = 0; j < CHUNKS; ++j) {
        const _Float16* p = xr + j * 32;

        // B layout (32x16, f16): lane(row,half) holds x[row][c + half*16 .. +15]
        v8h blo = *(const v8h*)(p + half * 16);
        v8h bhi = *(const v8h*)(p + half * 16 + 8);
        // A layout (16x32, f16): lanes 0-15: K 0..15; lanes 16-31: K 16..31
        v8h alo = *(const v8h*)(p + half * 16);
        v8h ahi = *(const v8h*)(p + half * 16 + 8);

        v16h a = __builtin_shufflevector(alo, ahi, 0,1,2,3,4,5,6,7,8,9,10,11,12,13,14,15);
        v16h b = __builtin_shufflevector(blo, bhi, 0,1,2,3,4,5,6,7,8,9,10,11,12,13,14,15);
        xb[j] = b;

        // D[m][n] += sum_k x[m][k]*x[n][k]; diagonal = per-row sum of squares.
        acc = __builtin_amdgcn_wmma_f32_16x16x32_f16(
                  /*neg_a=*/false, a, /*neg_b=*/false, b,
                  /*c_mod=*/(short)0, acc, /*reuse_a=*/false, /*reuse_b=*/false);
    }

    // Extract diag(acc): C/D layout: lanes 0-15 -> (M=v, N=lane),
    // lanes 16-31 -> (M=v+8, N=lane-16). (m,m): m<8 at lane m vgpr m;
    // m>=8 at lane m+16 vgpr m-8.
#pragma unroll
    for (int v = 0; v < 8; ++v) {
        if (lane == v)      atomicAdd(&ssum[v],     acc[v]);   // row v
        if (lane == v + 24) atomicAdd(&ssum[v + 8], acc[v]);   // row v+8
    }
    __syncthreads();

    // Per-row scale: rsqrt(mean(x^2) + eps), hardware rsqrt + 1 Newton step.
    const float var = ssum[row] * (1.0f / MEAN_DIM) + EPS;
    float s = rsqrtf(var);
    s = s * (1.5f - 0.5f * var * s * s);

    // Scale pass from registers (no global re-read); non-temporal 32B stores.
    _Float16* po = O + rowbase + (size_t)wave * K_PER_WAVE;
#pragma unroll
    for (int j = 0; j < CHUNKS; ++j) {
        v16h xv = xb[j];
        v16h o;
#pragma unroll
        for (int i = 0; i < 16; ++i)
            o[i] = (_Float16)((float)xv[i] * s);
        __builtin_nontemporal_store(o, (v16h*)(po + j * 32 + half * 16));
    }
}

extern "C" void kernel_launch(void* const* d_in, const int* in_sizes, int n_in,
                              void* d_out, int out_size, void* d_ws, size_t ws_size,
                              hipStream_t stream) {
    const _Float16* x = (const _Float16*)d_in[0];
    _Float16* out     = (_Float16*)d_out;

    const int rows   = in_sizes[0] / MEAN_DIM;        // 32768
    const int blocks = rows / ROWS_PER_BLOCK;         // 2048

    hipLaunchKernelGGL(rmsnorm_wmma_kernel, dim3(blocks), dim3(THREADS), 0, stream,
                       x, out);
}
